// TISABiasedScanAttention_23811298689322
// MI455X (gfx1250) — matrix operand order repaired and checked by reading, not compiled, not run on hardware
//
#include <hip/hip_runtime.h>

typedef __attribute__((ext_vector_type(16))) _Float16 v16h;
typedef __attribute__((ext_vector_type(2)))  __fp16   v2fp16;
typedef __attribute__((ext_vector_type(8)))  float    v8f;

#define B_  2
#define L_  2048
#define H_  8
#define D_  64
#define S_  2
#define F_  5
#define QT  64          // Q rows per block
#define KT  32          // keys per iteration
#define NWAVE 4
#define NTHREADS 128    // 4 wave32

#define LOG2E 1.4426950408889634f

__device__ __forceinline__ v8f wmma_f16(v16h a, v16h b, v8f c) {
    // D = A(16x32 f16) * B(32x16 f16) + C(16x16 f32)
    return __builtin_amdgcn_wmma_f32_16x16x32_f16(
        /*neg_a=*/false, a, /*neg_b=*/false, b,
        /*c_mod=*/(short)0, c, /*reuse_a=*/false, /*reuse_b=*/false);
}

// B-fragment: 16 contiguous f16 (32 bytes) -> 2x ds_load_b128
__device__ __forceinline__ v16h load_frag_b(const _Float16* p) {
    v16h r;
#pragma unroll
    for (int j = 0; j < 16; ++j) r[j] = p[j];
    return r;
}

// A-fragment: 16 contiguous f16 at p[0..15] (K = 16*hi + j)
__device__ __forceinline__ v16h load_frag_a(const _Float16* p) {
    v16h r;
#pragma unroll
    for (int j = 0; j < 16; ++j) r[j] = p[j];
    return r;
}

// v_cvt_pk_rtz_f16_f32 pair -> u32
__device__ __forceinline__ unsigned int pk_f16(float a, float b) {
    union { v2fp16 h; unsigned int u; } pk;
    pk.h = __builtin_amdgcn_cvt_pkrtz(a, b);
    return pk.u;
}

__global__ __launch_bounds__(NTHREADS)
void tisa_flash_attn(const float* __restrict__ qs,
                     const float* __restrict__ ks,
                     const float* __restrict__ vs,
                     const float* __restrict__ qs_s,
                     const float* __restrict__ ks_s,
                     const float* __restrict__ pa,
                     const float* __restrict__ pb,
                     const float* __restrict__ pc,
                     float* __restrict__ out)
{
    __shared__ _Float16 Qlds[QT][D_];          // 8 KB, scale pre-applied
    __shared__ _Float16 Klds[KT][D_];          // 4 KB, row-major (key, dim)
    __shared__ _Float16 Vt  [D_][KT];          // 4 KB, transposed (dim, key)
    __shared__ _Float16 Plds[NWAVE][16][KT];   // 4 KB, per-wave P buffer
    __shared__ float    qss [QT][S_];
    __shared__ float    kss [KT][S_];

    const int tid  = threadIdx.x;
    const int lane = tid & 31;
    const int wv   = tid >> 5;
    const int n    = lane & 15;   // N / column-in-tile index
    const int hi   = lane >> 4;   // half-wave select

    const int qtile = blockIdx.x;
    const int h     = blockIdx.y;
    const int bb    = blockIdx.z;
    const int qbase = qtile * QT;

    const float scale = 0.125f;   // 1/sqrt(D=64)

    // Per-head TISA bias parameters (uniform -> scalar loads).
    // Fold log2(e) into the Gaussian width so each term uses raw v_exp_f32.
    float af[F_], nbl[F_], cf[F_];
#pragma unroll
    for (int f = 0; f < F_; ++f) {
        af[f]  = pa[h * F_ + f];
        nbl[f] = -fabsf(pb[h * F_ + f]) * LOG2E;
        cf[f]  = pc[h * F_ + f];
    }

    // ---- Load Q tile (scaled, f32 -> f16 via v_cvt_pk_rtz) ----
#pragma unroll
    for (int e = 0; e < (QT * D_) / (4 * NTHREADS); ++e) {   // 8 float4 / thread
        const int lin = (e * NTHREADS + tid) * 4;
        const int row = lin >> 6, col = lin & 63;
        const float4 v = *(const float4*)&qs[(((size_t)bb * L_ + qbase + row) * H_ + h) * D_ + col];
        uint2 pk;
        pk.x = pk_f16(v.x * scale, v.y * scale);
        pk.y = pk_f16(v.z * scale, v.w * scale);
        *(uint2*)&Qlds[row][col] = pk;           // ds_store_b64
    }
    if (tid < QT * S_) {   // 128 coord floats
        const int row = tid >> 1, s = tid & 1;
        qss[row][s] = qs_s[((size_t)bb * L_ + qbase + row) * S_ + s];
    }
    __syncthreads();

    // ---- Q A-fragments (persist whole kernel): k-step 0 = dims 0..31, 1 = dims 32..63
    const v16h aq0 = load_frag_a(&Qlds[wv * 16 + n][ 0 + hi * 16]);
    const v16h aq1 = load_frag_a(&Qlds[wv * 16 + n][32 + hi * 16]);

    // qs_s for this lane's 8 score rows (row m = hi*8 + r)
    float q0r[8], q1r[8];
#pragma unroll
    for (int r = 0; r < 8; ++r) {
        const int m = wv * 16 + hi * 8 + r;
        q0r[r] = qss[m][0];
        q1r[r] = qss[m][1];
    }

    // ---- Online softmax state + output accumulators (16 x 64 per wave) ----
    float m_run[8], l_run[8];
#pragma unroll
    for (int r = 0; r < 8; ++r) { m_run[r] = -3.0e38f; l_run[r] = 0.0f; }
    v8f o0 = {}, o1 = {}, o2 = {}, o3 = {};

    for (int kb = 0; kb < L_ / KT; ++kb) {
        const int kbase = kb * KT;

        // ---- Cooperative K tile load (row-major f16) ----
#pragma unroll
        for (int e = 0; e < (KT * D_) / (4 * NTHREADS); ++e) {  // 4 float4 / thread
            const int lin = (e * NTHREADS + tid) * 4;
            const int row = lin >> 6, col = lin & 63;
            const float4 v = *(const float4*)&ks[(((size_t)bb * L_ + kbase + row) * H_ + h) * D_ + col];
            uint2 pk;
            pk.x = pk_f16(v.x, v.y);
            pk.y = pk_f16(v.z, v.w);
            *(uint2*)&Klds[row][col] = pk;       // ds_store_b64
        }
        // ---- Cooperative V tile load, transposed into Vt[dim][key] ----
#pragma unroll
        for (int e = 0; e < 2; ++e) {
            const int u  = e * NTHREADS + tid;   // 256 units: 16 key-pairs x 16 dim-groups
            const int kp = u & 15;               // key pair (2 keys)
            const int dg = u >> 4;               // dim group (4 dims)
            const float4 va = *(const float4*)&vs[(((size_t)bb * L_ + kbase + 2 * kp + 0) * H_ + h) * D_ + 4 * dg];
            const float4 vb = *(const float4*)&vs[(((size_t)bb * L_ + kbase + 2 * kp + 1) * H_ + h) * D_ + 4 * dg];
            const float ax[4] = {va.x, va.y, va.z, va.w};
            const float bx[4] = {vb.x, vb.y, vb.z, vb.w};
#pragma unroll
            for (int i = 0; i < 4; ++i) {
                // pack (key 2kp, key 2kp+1) for one dim -> one b32 store
                *(unsigned int*)&Vt[4 * dg + i][2 * kp] = pk_f16(ax[i], bx[i]);
            }
        }
        if (tid < KT * S_) {   // 64 coord floats
            const int row = tid >> 1, s = tid & 1;
            kss[row][s] = ks_s[((size_t)bb * L_ + kbase + row) * S_ + s];
        }
        __syncthreads();

        // Prefetch next K tile rows (global_prefetch_b8)
        if (kb + 1 < L_ / KT && tid < KT) {
            __builtin_prefetch(&ks[(((size_t)bb * L_ + kbase + KT + tid) * H_ + h) * D_], 0, 0);
        }

        // ---- S = Q K^T : two 16x16 C tiles (keys 0-15, 16-31) ----
        v8f c0 = {}, c1 = {};
        {
            v16h bk;
            bk = load_frag_b(&Klds[ 0 + n][ 0 + hi * 16]); c0 = wmma_f16(aq0, bk, c0);
            bk = load_frag_b(&Klds[ 0 + n][32 + hi * 16]); c0 = wmma_f16(aq1, bk, c0);
            bk = load_frag_b(&Klds[16 + n][ 0 + hi * 16]); c1 = wmma_f16(aq0, bk, c1);
            bk = load_frag_b(&Klds[16 + n][32 + hi * 16]); c1 = wmma_f16(aq1, bk, c1);
        }

        // ---- Bias + online softmax (row m = hi*8+r spans 16 lanes at vgpr r) ----
        const float ka0 = kss[n][0],      ka1 = kss[n][1];
        const float kb0 = kss[n + 16][0], kb1 = kss[n + 16][1];
        float alpha[8];
#pragma unroll
        for (int r = 0; r < 8; ++r) {
            float dx = q0r[r] - ka0, dy = q1r[r] - ka1;
            const float d0 = __builtin_amdgcn_sqrtf(dx * dx + dy * dy);   // raw v_sqrt_f32
            dx = q0r[r] - kb0; dy = q1r[r] - kb1;
            const float d1 = __builtin_amdgcn_sqrtf(dx * dx + dy * dy);
            float bias0 = 0.0f, bias1 = 0.0f;
#pragma unroll
            for (int f = 0; f < F_; ++f) {
                const float t0 = d0 - cf[f];
                const float t1 = d1 - cf[f];
                bias0 += af[f] * __builtin_amdgcn_exp2f(nbl[f] * t0 * t0); // raw v_exp_f32
                bias1 += af[f] * __builtin_amdgcn_exp2f(nbl[f] * t1 * t1);
            }
            const float s0 = c0[r] + bias0;
            const float s1 = c1[r] + bias1;

            float mx = fmaxf(s0, s1);
            mx = fmaxf(mx, __shfl_xor(mx, 1, 16));
            mx = fmaxf(mx, __shfl_xor(mx, 2, 16));
            mx = fmaxf(mx, __shfl_xor(mx, 4, 16));
            mx = fmaxf(mx, __shfl_xor(mx, 8, 16));

            const float mnew = fmaxf(m_run[r], mx);
            const float al   = __builtin_amdgcn_exp2f((m_run[r] - mnew) * LOG2E);
            m_run[r] = mnew;
            alpha[r] = al;

            const float p0 = __builtin_amdgcn_exp2f((s0 - mnew) * LOG2E);
            const float p1 = __builtin_amdgcn_exp2f((s1 - mnew) * LOG2E);
            float rs = p0 + p1;
            rs += __shfl_xor(rs, 1, 16);
            rs += __shfl_xor(rs, 2, 16);
            rs += __shfl_xor(rs, 4, 16);
            rs += __shfl_xor(rs, 8, 16);
            l_run[r] = l_run[r] * al + rs;

            // P -> per-wave LDS (C-layout -> A-layout staging)
            Plds[wv][hi * 8 + r][n]      = (_Float16)p0;
            Plds[wv][hi * 8 + r][n + 16] = (_Float16)p1;
        }

        // LDS ops are per-wave in order; explicit split-counter wait before readback
        asm volatile("s_wait_dscnt 0" ::: "memory");

        // Rescale accumulators by alpha (same (lane,r) -> row mapping as C tiles)
#pragma unroll
        for (int r = 0; r < 8; ++r) {
            o0[r] *= alpha[r]; o1[r] *= alpha[r];
            o2[r] *= alpha[r]; o3[r] *= alpha[r];
        }

        // ---- O += P V : A = P (16x32), B = Vt rows (contiguous keys) ----
        {
            const v16h ap = load_frag_a(&Plds[wv][n][hi * 16]);
            v16h bv;
            bv = load_frag_b(&Vt[ 0 + n][hi * 16]); o0 = wmma_f16(ap, bv, o0);
            bv = load_frag_b(&Vt[16 + n][hi * 16]); o1 = wmma_f16(ap, bv, o1);
            bv = load_frag_b(&Vt[32 + n][hi * 16]); o2 = wmma_f16(ap, bv, o2);
            bv = load_frag_b(&Vt[48 + n][hi * 16]); o3 = wmma_f16(ap, bv, o3);
        }
        __syncthreads();   // protect Klds/Vt/kss before next iteration overwrites
    }

    // ---- Normalize (v_rcp_f32) and store ----
#pragma unroll
    for (int r = 0; r < 8; ++r) {
        const float inv = __builtin_amdgcn_rcpf(l_run[r]);
        const int q = qbase + wv * 16 + hi * 8 + r;
        const size_t base = (((size_t)bb * L_ + q) * H_ + h) * D_;
        out[base +  0 + n] = o0[r] * inv;
        out[base + 16 + n] = o1[r] * inv;
        out[base + 32 + n] = o2[r] * inv;
        out[base + 48 + n] = o3[r] * inv;
    }
}

extern "C" void kernel_launch(void* const* d_in, const int* in_sizes, int n_in,
                              void* d_out, int out_size, void* d_ws, size_t ws_size,
                              hipStream_t stream) {
    (void)in_sizes; (void)n_in; (void)d_ws; (void)ws_size; (void)out_size;
    const float* qs   = (const float*)d_in[0];
    const float* ks   = (const float*)d_in[1];
    const float* vs   = (const float*)d_in[2];
    const float* qs_s = (const float*)d_in[3];
    const float* ks_s = (const float*)d_in[4];
    const float* pa   = (const float*)d_in[5];
    const float* pb   = (const float*)d_in[6];
    const float* pc   = (const float*)d_in[7];
    float* out = (float*)d_out;

    dim3 grid(L_ / QT, H_, B_);
    tisa_flash_attn<<<grid, NTHREADS, 0, stream>>>(qs, ks, vs, qs_s, ks_s, pa, pb, pc, out);
}
